// MultiHeadAttention_40630390621144
// MI455X (gfx1250) — matrix operand order repaired and checked by reading, not compiled, not run on hardware
//
#include <hip/hip_runtime.h>

typedef __bf16 bf16_t;
typedef __attribute__((ext_vector_type(16))) __bf16 v16bf;
typedef __attribute__((ext_vector_type(8)))  __bf16 v8bf;
typedef __attribute__((ext_vector_type(8)))  float  v8f;

#define D_MODEL 1024
#define SEQ     2048
#define BATCH   2
#define HEADS   16
#define HDIM    64
#define TOKENS  (BATCH * SEQ)   // 4096

// ---------------------------------------------------------------------------
// WMMA helpers (CDNA5 wave32, 16x16x32 bf16 -> f32)
// ---------------------------------------------------------------------------
__device__ __forceinline__ v8f wmma_bf(v16bf a, v16bf b, v8f c) {
  return __builtin_amdgcn_wmma_f32_16x16x32_bf16(false, a, false, b, (short)0, c,
                                                 false, false);
}

// A-fragment (16x32 bf16, row-major source, ld = lda elements).
// ISA layout: lane l holds row M=l&15; kHalf=l>>4.
// elements 0..7  : K = k0 + kHalf*8 + e
// elements 8..15 : K = k0 + 16 + kHalf*8 + (e-8)
__device__ __forceinline__ v16bf load_a_frag(const bf16_t* base, int lda, int row,
                                             int k0, int kHalf) {
  const bf16_t* p = base + row * lda + k0 + kHalf * 8;
  v8bf lo = *(const v8bf*)(p);
  v8bf hi = *(const v8bf*)(p + 16);
  v16bf a;
#pragma unroll
  for (int i = 0; i < 8; ++i) { a[i] = lo[i]; a[i + 8] = hi[i]; }
  return a;
}

// B-fragment (32x16 bf16) from a TRANSPOSED B (Bt[N][K] row-major, ld = ldb).
// ISA layout: lane n (0..15) holds column N=n; kHalf=l>>4.
// elements 0..7 : K = k0 + kHalf*8 + e; elements 8..15 : K = k0 + 16 + kHalf*8 + (e-8)
__device__ __forceinline__ v16bf load_b_frag(const bf16_t* bt, int ldb, int col,
                                             int k0, int kHalf) {
  const bf16_t* p = bt + col * ldb + k0 + kHalf * 8;
  v8bf lo = *(const v8bf*)(p);
  v8bf hi = *(const v8bf*)(p + 16);
  v16bf b;
#pragma unroll
  for (int i = 0; i < 8; ++i) { b[i] = lo[i]; b[i + 8] = hi[i]; }
  return b;
}

// ---------------------------------------------------------------------------
// Precision conversion kernels
// ---------------------------------------------------------------------------
__global__ void cvt_f32_to_bf16(const float* __restrict__ in,
                                bf16_t* __restrict__ out, int n) {
  int i = blockIdx.x * blockDim.x + threadIdx.x;
  if (i < n) out[i] = (bf16_t)in[i];
}

// Four W[K][N] f32 (row-major)  ->  Wt[N][K] bf16 (row-major), z selects matrix.
__global__ void transpose_cvt4(const float* __restrict__ w0, const float* __restrict__ w1,
                               const float* __restrict__ w2, const float* __restrict__ w3,
                               bf16_t* __restrict__ o0, bf16_t* __restrict__ o1,
                               bf16_t* __restrict__ o2, bf16_t* __restrict__ o3) {
  const float* src;
  bf16_t* dst;
  switch (blockIdx.z) {
    case 0:  src = w0; dst = o0; break;
    case 1:  src = w1; dst = o1; break;
    case 2:  src = w2; dst = o2; break;
    default: src = w3; dst = o3; break;
  }
  const int n = blockIdx.x * blockDim.x + threadIdx.x;
  const int k = blockIdx.y;
  dst[n * D_MODEL + k] = (bf16_t)src[k * D_MODEL + n];
}

// ---------------------------------------------------------------------------
// Fused QKV projection.  C = xb[4096,1024] @ W  (via Wt[N][K]).
// grid: (64 mGroups, 4 colQuads, 3 {q,k,v}); block 128 (4 waves).
// Each wave: 64 rows x 64 cols (4x4 accumulators) -> 16 WMMA per 16 b128 loads.
// Q,K stored [b,h,n,64]; V stored transposed [b,h,64,n].
// ---------------------------------------------------------------------------
__global__ __launch_bounds__(128) void qkv_gemm(
    const bf16_t* __restrict__ xb, const bf16_t* __restrict__ wqt,
    const bf16_t* __restrict__ wkt, const bf16_t* __restrict__ wvt,
    bf16_t* __restrict__ Qb, bf16_t* __restrict__ Kb, bf16_t* __restrict__ Vt) {
  const int lane  = threadIdx.x & 31;
  const int wave  = threadIdx.x >> 5;
  const int kHalf = lane >> 4;
  const int ln    = lane & 15;
  const int mBase = blockIdx.x * 64;
  const int col0  = (blockIdx.y * 4 + wave) * 64;
  const int which = blockIdx.z;
  const bf16_t* wt = (which == 0) ? wqt : (which == 1) ? wkt : wvt;

  v8f acc[4][4];
#pragma unroll
  for (int mt = 0; mt < 4; ++mt)
#pragma unroll
    for (int nt = 0; nt < 4; ++nt)
#pragma unroll
      for (int i = 0; i < 8; ++i) acc[mt][nt][i] = 0.0f;

  for (int k0 = 0; k0 < D_MODEL; k0 += 32) {
    v16bf a[4];
#pragma unroll
    for (int mt = 0; mt < 4; ++mt)
      a[mt] = load_a_frag(xb, D_MODEL, mBase + mt * 16 + ln, k0, kHalf);
#pragma unroll
    for (int nt = 0; nt < 4; ++nt) {
      v16bf b = load_b_frag(wt, D_MODEL, col0 + nt * 16 + ln, k0, kHalf);
#pragma unroll
      for (int mt = 0; mt < 4; ++mt) acc[mt][nt] = wmma_bf(a[mt], b, acc[mt][nt]);
    }
  }

#pragma unroll
  for (int nt = 0; nt < 4; ++nt) {
    const int c = col0 + nt * 16 + ln;
    const int h = c >> 6, d = c & 63;
#pragma unroll
    for (int mt = 0; mt < 4; ++mt)
#pragma unroll
      for (int r = 0; r < 8; ++r) {
        const int tok = mBase + mt * 16 + r + 8 * kHalf;
        const int b   = tok >> 11, nI = tok & (SEQ - 1);
        const bf16_t v = (bf16_t)acc[mt][nt][r];
        if (which == 0)      Qb[(((b * HEADS + h) * SEQ) + nI) * HDIM + d] = v;
        else if (which == 1) Kb[(((b * HEADS + h) * SEQ) + nI) * HDIM + d] = v;
        else                 Vt[(((b * HEADS + h) * HDIM) + d) * SEQ + nI] = v;
      }
  }
}

// ---------------------------------------------------------------------------
// Flash attention (causal, online softmax). One wave = 32 query rows x 1 head;
// K/V fragments loaded once per 32-key strip, reused by both 16-row q-tiles.
// grid: (16 qOcts, 16 heads, 2 batch); block 128 (4 waves).
// ---------------------------------------------------------------------------
__global__ __launch_bounds__(128) void flash_attn(
    const bf16_t* __restrict__ Qb, const bf16_t* __restrict__ Kb,
    const bf16_t* __restrict__ Vt, bf16_t* __restrict__ ctx) {
  __shared__ bf16_t ldsP[4][2][16 * 32];

  const int lane  = threadIdx.x & 31;
  const int wave  = threadIdx.x >> 5;
  const int kHalf = lane >> 4;
  const int ln    = lane & 15;
  const int h     = blockIdx.y;
  const int b     = blockIdx.z;
  const int qBase = (blockIdx.x * 4 + wave) * 32;   // multiple of 32

  const bf16_t* Qh = Qb + (size_t)(b * HEADS + h) * SEQ * HDIM;  // [n,64]
  const bf16_t* Kh = Kb + (size_t)(b * HEADS + h) * SEQ * HDIM;  // [n,64] == Bt for Q@K^T
  const bf16_t* Vh = Vt + (size_t)(b * HEADS + h) * HDIM * SEQ;  // [64,n] == Bt for P@V

  // Q fragments: two q-tiles x two 16x32 A-fragments
  v16bf qa[2][2];
#pragma unroll
  for (int qt = 0; qt < 2; ++qt) {
    qa[qt][0] = load_a_frag(Qh, HDIM, qBase + qt * 16 + ln, 0, kHalf);
    qa[qt][1] = load_a_frag(Qh, HDIM, qBase + qt * 16 + ln, 32, kHalf);
  }

  float m[2][8], l[2][8];
  v8f o[2][4];
#pragma unroll
  for (int qt = 0; qt < 2; ++qt) {
#pragma unroll
    for (int r = 0; r < 8; ++r) { m[qt][r] = -3.0e38f; l[qt][r] = 0.0f; }
#pragma unroll
    for (int nt = 0; nt < 4; ++nt)
#pragma unroll
      for (int i = 0; i < 8; ++i) o[qt][nt][i] = 0.0f;
  }

  const float scale = 0.125f;  // 1/sqrt(64)

  for (int kb = 0; kb <= qBase; kb += 32) {
    // ---- K fragments, shared by both q-tiles ----
    v16bf k0a = load_b_frag(Kh, HDIM, kb + ln, 0, kHalf);
    v16bf k0b = load_b_frag(Kh, HDIM, kb + ln, 32, kHalf);
    v16bf k1a = load_b_frag(Kh, HDIM, kb + 16 + ln, 0, kHalf);
    v16bf k1b = load_b_frag(Kh, HDIM, kb + 16 + ln, 32, kHalf);
    const bool needMask = (kb + 31 > qBase);  // true only for the diagonal strip

#pragma unroll
    for (int qt = 0; qt < 2; ++qt) {
      // ---- scores: two 16x16 tiles over keys [kb, kb+32) ----
      v8f s0, s1;
#pragma unroll
      for (int i = 0; i < 8; ++i) { s0[i] = 0.0f; s1[i] = 0.0f; }
      s0 = wmma_bf(qa[qt][0], k0a, s0);
      s0 = wmma_bf(qa[qt][1], k0b, s0);
      s1 = wmma_bf(qa[qt][0], k1a, s1);
      s1 = wmma_bf(qa[qt][1], k1b, s1);

#pragma unroll
      for (int r = 0; r < 8; ++r) { s0[r] *= scale; s1[r] *= scale; }
      if (needMask) {
#pragma unroll
        for (int r = 0; r < 8; ++r) {
          const int qRow = qBase + qt * 16 + r + 8 * kHalf;
          if (kb + ln > qRow)      s0[r] = -1.0e30f;
          if (kb + 16 + ln > qRow) s1[r] = -1.0e30f;
        }
      }
      // ---- online softmax ----
      float corr[8];
#pragma unroll
      for (int r = 0; r < 8; ++r) {
        float v = fmaxf(s0[r], s1[r]);
#pragma unroll
        for (int off = 8; off; off >>= 1) v = fmaxf(v, __shfl_xor(v, off, 16));
        const float mn = fmaxf(m[qt][r], v);
        corr[r] = __expf(m[qt][r] - mn);
        m[qt][r] = mn;
      }
#pragma unroll
      for (int r = 0; r < 8; ++r) {
        const float p0 = __expf(s0[r] - m[qt][r]);
        const float p1 = __expf(s1[r] - m[qt][r]);
        s0[r] = p0; s1[r] = p1;
        float t = p0 + p1;
#pragma unroll
        for (int off = 8; off; off >>= 1) t += __shfl_xor(t, off, 16);
        l[qt][r] = l[qt][r] * corr[r] + t;
      }
#pragma unroll
      for (int nt = 0; nt < 4; ++nt)
#pragma unroll
        for (int r = 0; r < 8; ++r) o[qt][nt][r] *= corr[r];

      // ---- P (C-layout) -> LDS (16x32 bf16 row-major) ----
      bf16_t* lp = ldsP[wave][qt];
#pragma unroll
      for (int r = 0; r < 8; ++r) {
        const int row = r + 8 * kHalf;
        lp[row * 32 + ln]      = (bf16_t)s0[r];
        lp[row * 32 + 16 + ln] = (bf16_t)s1[r];
      }
    }
    asm volatile("s_wait_dscnt 0" ::: "memory");
    v16bf pa0 = load_a_frag(ldsP[wave][0], 32, ln, 0, kHalf);
    v16bf pa1 = load_a_frag(ldsP[wave][1], 32, ln, 0, kHalf);

    // ---- O += P @ V  (V fragments shared by both q-tiles) ----
#pragma unroll
    for (int nt = 0; nt < 4; ++nt) {
      v16bf vb = load_b_frag(Vh, SEQ, nt * 16 + ln, kb, kHalf);
      o[0][nt] = wmma_bf(pa0, vb, o[0][nt]);
      o[1][nt] = wmma_bf(pa1, vb, o[1][nt]);
    }
  }

  // ---- epilogue: ctx[b*SEQ+q, h*64 + c] = o / l ----
#pragma unroll
  for (int qt = 0; qt < 2; ++qt) {
    float inv[8];
#pragma unroll
    for (int r = 0; r < 8; ++r) inv[r] = 1.0f / l[qt][r];
#pragma unroll
    for (int nt = 0; nt < 4; ++nt) {
      const int c = h * HDIM + nt * 16 + ln;
#pragma unroll
      for (int r = 0; r < 8; ++r) {
        const int tok = b * SEQ + qBase + qt * 16 + r + 8 * kHalf;
        ctx[(size_t)tok * D_MODEL + c] = (bf16_t)(o[qt][nt][r] * inv[r]);
      }
    }
  }
}

// ---------------------------------------------------------------------------
// Output projection: out = ctx @ Wo + bo   (f32 output)
// grid: (64 mGroups, 4 colQuads); block 128 (4 waves), wave = 64x64 tile.
// ---------------------------------------------------------------------------
__global__ __launch_bounds__(128) void out_proj(
    const bf16_t* __restrict__ ctxb, const bf16_t* __restrict__ wot,
    const float* __restrict__ bo, float* __restrict__ out) {
  const int lane  = threadIdx.x & 31;
  const int wave  = threadIdx.x >> 5;
  const int kHalf = lane >> 4;
  const int ln    = lane & 15;
  const int mBase = blockIdx.x * 64;
  const int col0  = (blockIdx.y * 4 + wave) * 64;

  v8f acc[4][4];
#pragma unroll
  for (int mt = 0; mt < 4; ++mt)
#pragma unroll
    for (int nt = 0; nt < 4; ++nt)
#pragma unroll
      for (int i = 0; i < 8; ++i) acc[mt][nt][i] = 0.0f;

  for (int k0 = 0; k0 < D_MODEL; k0 += 32) {
    v16bf a[4];
#pragma unroll
    for (int mt = 0; mt < 4; ++mt)
      a[mt] = load_a_frag(ctxb, D_MODEL, mBase + mt * 16 + ln, k0, kHalf);
#pragma unroll
    for (int nt = 0; nt < 4; ++nt) {
      v16bf b = load_b_frag(wot, D_MODEL, col0 + nt * 16 + ln, k0, kHalf);
#pragma unroll
      for (int mt = 0; mt < 4; ++mt) acc[mt][nt] = wmma_bf(a[mt], b, acc[mt][nt]);
    }
  }

#pragma unroll
  for (int nt = 0; nt < 4; ++nt) {
    const int c = col0 + nt * 16 + ln;
    const float bias = bo[c];
#pragma unroll
    for (int mt = 0; mt < 4; ++mt)
#pragma unroll
      for (int r = 0; r < 8; ++r) {
        const int tok = mBase + mt * 16 + r + 8 * kHalf;
        out[(size_t)tok * D_MODEL + c] = acc[mt][nt][r] + bias;
      }
  }
}

// ---------------------------------------------------------------------------
// Host launcher
// ---------------------------------------------------------------------------
extern "C" void kernel_launch(void* const* d_in, const int* in_sizes, int n_in,
                              void* d_out, int out_size, void* d_ws, size_t ws_size,
                              hipStream_t stream) {
  const float* x  = (const float*)d_in[0];
  const float* Wq = (const float*)d_in[1];
  const float* Wk = (const float*)d_in[2];
  const float* Wv = (const float*)d_in[3];
  const float* Wo = (const float*)d_in[4];
  const float* bo = (const float*)d_in[5];
  float* out = (float*)d_out;

  const size_t XE = (size_t)TOKENS * D_MODEL;            // 4,194,304
  const size_t WE = (size_t)D_MODEL * D_MODEL;           // 1,048,576
  const size_t HE = (size_t)BATCH * HEADS * SEQ * HDIM;  // 4,194,304

  bf16_t* xb   = (bf16_t*)d_ws;
  bf16_t* wqt  = xb + XE;
  bf16_t* wkt  = wqt + WE;
  bf16_t* wvt  = wkt + WE;
  bf16_t* wot  = wvt + WE;
  bf16_t* Qb   = wot + WE;
  bf16_t* Kb   = Qb + HE;
  bf16_t* Vt   = Kb + HE;
  bf16_t* ctxb = Vt + HE;

  // 1) fp32 -> bf16 conversions (x) and transposed weights (all four in one grid)
  cvt_f32_to_bf16<<<(int)(XE / 256), 256, 0, stream>>>(x, xb, (int)XE);
  transpose_cvt4<<<dim3(D_MODEL / 256, D_MODEL, 4), 256, 0, stream>>>(
      Wq, Wk, Wv, Wo, wqt, wkt, wvt, wot);

  // 2) fused QKV projection (grid.z selects Q/K/V)
  qkv_gemm<<<dim3(TOKENS / 64, 4, 3), 128, 0, stream>>>(xb, wqt, wkt, wvt, Qb, Kb, Vt);

  // 3) causal flash attention (32 query rows per wave)
  flash_attn<<<dim3(SEQ / 128, HEADS, BATCH), 128, 0, stream>>>(Qb, Kb, Vt, ctxb);

  // 4) output projection + bias
  out_proj<<<dim3(TOKENS / 64, 4, 1), 128, 0, stream>>>(ctxb, wot, bo, out);
}
